// DN4_10668698763884
// MI455X (gfx1250) — matrix operand order repaired and checked by reading, not compiled, not run on hardware
//
#include <hip/hip_runtime.h>
#include <hip/hip_bf16.h>

typedef _Float16 f16;
typedef __attribute__((ext_vector_type(16))) _Float16 v16h;
typedef __attribute__((ext_vector_type(8)))  float    v8f;

union Frag16 { v16h h; uint4 u[2]; };

#define WMMA_F16(A, B, C) \
  __builtin_amdgcn_wmma_f32_16x16x32_f16(false, (A), false, (B), (short)0, (C), false, false)

#define NIMG   82
#define NQIMG  32
#define BSZ    2
#define NQn    16
#define WAY    5
#define SHOT   5
#define H0     84
#define Hp1    42
#define Hp2    21
#define Lloc   441      /* 21*21 */
#define Mtot   2205     /* SHOT*441 */
#define DCH    64

#define STATS_BLOCKS 128

/* workspace layout (bytes) */
#define OFF_RAW   0ul           /* f16 NHWC raw conv out, max 82*84*84*64*2 = 74,059,776 */
#define OFF_ACT   74059776ul    /* f16 NHWC activations, max 82*42*42*64*2 = 18,514,944 */
#define OFF_FEAT  92574720ul    /* f16 normalized descriptors 82*441*64*2 = 4,628,736 */
#define OFF_WT    97203456ul    /* f16 reordered weights 3 * 64*576*2 */
#define OFF_PART  97424640ul    /* f32 BN partials 2 * 128*64*2 */
#define OFF_SS    97555712ul    /* f32 BN scale/shift [2][64][2] */

__device__ __forceinline__ float lrelu(float v) { return v >= 0.f ? v : 0.2f * v; }

__device__ __forceinline__ void top3_insert(float v, float& t0, float& t1, float& t2) {
  if (v > t2) {
    if (v > t0)      { t2 = t1; t1 = t0; t0 = v; }
    else if (v > t1) { t2 = t1; t1 = v; }
    else             { t2 = v; }
  }
}

/* ---- weight reorder: W[co][ci][3][3] f32 -> Wt[co][(ky*3+kx)*64+ci] f16 ---- */
__global__ void k_wprep(const float* __restrict__ W, f16* __restrict__ Wt) {
  int idx = blockIdx.x * blockDim.x + threadIdx.x;
  if (idx >= 64 * 576) return;
  int co = idx / 576, kt = idx % 576;
  int w9 = kt / 64, ci = kt % 64;
  Wt[idx] = (f16)W[co * 576 + ci * 9 + w9];
}

/* ---- conv1: 3->64, 84x84, direct fp32, output f16 NHWC ---- */
__global__ void k_conv1(const float* __restrict__ q, const float* __restrict__ s,
                        const float* __restrict__ W1, f16* __restrict__ raw) {
  __shared__ float wsh[64 * 27];
  for (int i = threadIdx.x; i < 64 * 27; i += blockDim.x) wsh[i] = W1[i];
  __syncthreads();
  int gid = blockIdx.x * blockDim.x + threadIdx.x;
  if (gid >= NIMG * H0 * H0) return;
  int img = gid / (H0 * H0), p = gid % (H0 * H0);
  int y = p / H0, x = p % H0;
  const float* base = (img < NQIMG) ? q + (long)img * 3 * H0 * H0
                                    : s + (long)(img - NQIMG) * 3 * H0 * H0;
  float xin[27];
#pragma unroll
  for (int ci = 0; ci < 3; ++ci)
#pragma unroll
    for (int dy = 0; dy < 3; ++dy)
#pragma unroll
      for (int dx = 0; dx < 3; ++dx) {
        int ys = y + dy - 1, xs = x + dx - 1;
        bool ok = (ys >= 0) && (ys < H0) && (xs >= 0) && (xs < H0);
        xin[ci * 9 + dy * 3 + dx] = ok ? base[ci * H0 * H0 + ys * H0 + xs] : 0.f;
      }
  union { f16 h[64]; uint4 u[8]; } oh;
#pragma unroll
  for (int co = 0; co < 64; ++co) {
    float a = 0.f;
#pragma unroll
    for (int j = 0; j < 27; ++j) a += xin[j] * wsh[co * 27 + j];
    oh.h[co] = (f16)a;
  }
  uint4* op = (uint4*)(raw + (long)gid * 64);
#pragma unroll
  for (int i = 0; i < 8; ++i) op[i] = oh.u[i];
}

/* ---- BN stats stage 1: per-block per-channel partial sums (deterministic) ---- */
__global__ void k_stats(const f16* __restrict__ raw, long pixBase, long nPix,
                        float* __restrict__ part) {
  int c = threadIdx.x & 63, row = threadIdx.x >> 6;
  float s = 0.f, s2 = 0.f;
  for (long pix = (long)blockIdx.x * 4 + row; pix < nPix; pix += (long)gridDim.x * 4) {
    float v = (float)raw[(pixBase + pix) * 64 + c];
    s += v; s2 += v * v;
  }
  __shared__ float sh0[256], sh1[256];
  sh0[threadIdx.x] = s; sh1[threadIdx.x] = s2;
  __syncthreads();
  if (threadIdx.x < 64) {
    s  = sh0[c] + sh0[c + 64] + sh0[c + 128] + sh0[c + 192];
    s2 = sh1[c] + sh1[c + 64] + sh1[c + 128] + sh1[c + 192];
    part[((long)blockIdx.x * 64 + c) * 2]     = s;
    part[((long)blockIdx.x * 64 + c) * 2 + 1] = s2;
  }
}

/* ---- BN stats stage 2: fixed-order reduce, scale/shift ---- */
__global__ void k_bnfin(const float* __restrict__ p0, const float* __restrict__ p1,
                        const float* __restrict__ gamma, const float* __restrict__ beta,
                        float* __restrict__ ss, long cnt0, long cnt1) {
  int t = threadIdx.x;
  if (t >= 128) return;
  int g = t >> 6, c = t & 63;
  const float* part = g ? p1 : p0;
  long cnt = g ? cnt1 : cnt0;
  float s = 0.f, s2 = 0.f;
  for (int i = 0; i < STATS_BLOCKS; ++i) {
    s  += part[(i * 64 + c) * 2];
    s2 += part[(i * 64 + c) * 2 + 1];
  }
  float inv = 1.f / (float)cnt;
  float mean = s * inv;
  float var = s2 * inv - mean * mean;
  float sc = gamma[c] * rsqrtf(var + 1e-5f);
  ss[(g * 64 + c) * 2]     = sc;
  ss[(g * 64 + c) * 2 + 1] = beta[c] - mean * sc;
}

/* ---- BN affine + LeakyReLU + 2x2 maxpool, f16 NHWC in/out ---- */
__global__ void k_normpool(const f16* __restrict__ raw, f16* __restrict__ act,
                           const float* __restrict__ ss, int Hin) {
  __shared__ float ssl[256];
  for (int i = threadIdx.x; i < 256; i += blockDim.x) ssl[i] = ss[i];
  __syncthreads();
  int Hout = Hin >> 1;
  int gid = blockIdx.x * blockDim.x + threadIdx.x;
  if (gid >= NIMG * Hout * Hout) return;
  int img = gid / (Hout * Hout), p = gid % (Hout * Hout);
  int yo = p / Hout, xo = p % Hout;
  int g = (img < NQIMG) ? 0 : 1;
  const f16* r0 = raw + (((long)img * Hin + 2 * yo) * Hin + 2 * xo) * 64;
  const f16* r2 = r0 + (long)Hin * 64;
  union { f16 h[64]; uint4 u[8]; } oh;
#pragma unroll 8
  for (int c = 0; c < 64; ++c) {
    float sc = ssl[(g * 64 + c) * 2], sh = ssl[(g * 64 + c) * 2 + 1];
    float v0 = lrelu(sc * (float)r0[c] + sh);
    float v1 = lrelu(sc * (float)r0[64 + c] + sh);
    float v2 = lrelu(sc * (float)r2[c] + sh);
    float v3 = lrelu(sc * (float)r2[64 + c] + sh);
    oh.h[c] = (f16)fmaxf(fmaxf(v0, v1), fmaxf(v2, v3));
  }
  uint4* op = (uint4*)(act + (long)gid * 64);
#pragma unroll
  for (int i = 0; i < 8; ++i) op[i] = oh.u[i];
}

/* ---- BN affine + LeakyReLU (no pool), 21x21 ---- */
__global__ void k_norm(const f16* __restrict__ raw, f16* __restrict__ act,
                       const float* __restrict__ ss) {
  __shared__ float ssl[256];
  for (int i = threadIdx.x; i < 256; i += blockDim.x) ssl[i] = ss[i];
  __syncthreads();
  int gid = blockIdx.x * blockDim.x + threadIdx.x;
  if (gid >= NIMG * Lloc) return;
  int img = gid / Lloc;
  int g = (img < NQIMG) ? 0 : 1;
  const f16* r = raw + (long)gid * 64;
  union { f16 h[64]; uint4 u[8]; } oh;
#pragma unroll 8
  for (int c = 0; c < 64; ++c) {
    float sc = ssl[(g * 64 + c) * 2], sh = ssl[(g * 64 + c) * 2 + 1];
    oh.h[c] = (f16)lrelu(sc * (float)r[c] + sh);
  }
  uint4* op = (uint4*)(act + (long)gid * 64);
#pragma unroll
  for (int i = 0; i < 8; ++i) op[i] = oh.u[i];
}

/* ---- layer-4 BN + LeakyReLU + per-descriptor L2 normalize ---- */
__global__ void k_norml2(const f16* __restrict__ raw, f16* __restrict__ feat,
                         const float* __restrict__ ss) {
  __shared__ float ssl[256];
  for (int i = threadIdx.x; i < 256; i += blockDim.x) ssl[i] = ss[i];
  __syncthreads();
  int gid = blockIdx.x * blockDim.x + threadIdx.x;
  if (gid >= NIMG * Lloc) return;
  int img = gid / Lloc;
  int g = (img < NQIMG) ? 0 : 1;
  const f16* r = raw + (long)gid * 64;
  float ssq = 0.f;
#pragma unroll 8
  for (int c = 0; c < 64; ++c) {
    float sc = ssl[(g * 64 + c) * 2], sh = ssl[(g * 64 + c) * 2 + 1];
    float v = lrelu(sc * (float)r[c] + sh);
    ssq += v * v;
  }
  float inv = 1.f / fmaxf(sqrtf(ssq), 1e-12f);
  union { f16 h[64]; uint4 u[8]; } oh;
#pragma unroll 8
  for (int c = 0; c < 64; ++c) {
    float sc = ssl[(g * 64 + c) * 2], sh = ssl[(g * 64 + c) * 2 + 1];
    oh.h[c] = (f16)(lrelu(sc * (float)r[c] + sh) * inv);
  }
  uint4* op = (uint4*)(feat + (long)gid * 64);
#pragma unroll
  for (int i = 0; i < 8; ++i) op[i] = oh.u[i];
}

/* ---- conv 64->64 3x3 pad1 via im2col WMMA f16, NHWC in/out, K=576 ----
   Weights staged global->LDS with GLOBAL_LOAD_ASYNC_TO_LDS_B128 (ASYNCcnt).
   kc loop fully unrolled: window indices constant-fold, center taps lose their
   bounds checks, B-fragment LDS reads become immediate-offset ds_load_b128.   */
__global__ void k_convw(const f16* __restrict__ act, const f16* __restrict__ wt,
                        f16* __restrict__ raw, int Hd) {
  extern __shared__ f16 wl[]; /* 64*576 halves = 73,728 B */
  {
    /* async copy: per-lane 16B global -> LDS, no VGPR round trip */
    unsigned ldsBase = (unsigned)(unsigned long long)(void*)wl;
    for (int i = threadIdx.x; i < (64 * 576) / 8; i += blockDim.x) {
      unsigned ldsoff = ldsBase + (unsigned)i * 16u;
      const f16* g = wt + (long)i * 8;
      asm volatile("global_load_async_to_lds_b128 %0, %1, off"
                   :: "v"(ldsoff), "v"(g) : "memory");
    }
    asm volatile("s_wait_asynccnt 0x0" ::: "memory");
  }
  __syncthreads();
  int P = Hd * Hd;
  int ntiles = (P + 15) >> 4;
  int wave = threadIdx.x >> 5, lane = threadIdx.x & 31;
  int tile = blockIdx.x * 8 + wave;
  if (tile >= ntiles) return;
  int img = blockIdx.y;
  int m = tile * 16 + (lane & 15);
  bool vm = m < P;
  int mm = vm ? m : 0;
  int y = mm / Hd, x = mm % Hd;
  int kb = (lane >> 4) * 8;   /* A K-seg base: {0,8}; seg1 = kb+16 */
  int db = (lane >> 4) * 8;   /* B K-seg base: {0,8}; seg1 = db+16 */
  const f16* wlane = wl + (lane & 15) * 576 + db;
  v8f acc[4] = {};
#pragma unroll
  for (int kc = 0; kc < 18; ++kc) {
    const int w9 = kc >> 1, cib = (kc & 1) * 32;
    const int ky = w9 / 3, kx = w9 % 3;
    int ys = y + ky - 1, xs = x + kx - 1;
    bool v = vm && (ys >= 0) && (ys < Hd) && (xs >= 0) && (xs < Hd);
    Frag16 a;
    if (v) {
      const f16* pb = act + (((long)img * Hd + ys) * Hd + xs) * 64 + cib;
      a.u[0] = *(const uint4*)(pb + kb);
      a.u[1] = *(const uint4*)(pb + kb + 16);
    } else {
      a.u[0] = make_uint4(0, 0, 0, 0);
      a.u[1] = make_uint4(0, 0, 0, 0);
    }
#pragma unroll
    for (int cot = 0; cot < 4; ++cot) {
      const uint4* bp = (const uint4*)(wlane + cot * 16 * 576 + kc * 32);
      Frag16 bf;
      bf.u[0] = bp[0]; bf.u[1] = bp[2];
      acc[cot] = WMMA_F16(a.h, bf.h, acc[cot]);
    }
  }
  int rb = (lane >> 4) * 8;
#pragma unroll
  for (int r = 0; r < 8; ++r) {
    int pr = tile * 16 + r + rb;
    if (pr < P) {
      f16* o = raw + ((long)img * P + pr) * 64 + (lane & 15);
      o[0]  = (f16)acc[0][r];
      o[16] = (f16)acc[1][r];
      o[32] = (f16)acc[2][r];
      o[48] = (f16)acc[3][r];
    }
  }
}

/* ---- fused cosine-similarity GEMM + streaming top-3 + sums ----
   Per (b,q,c): D-tile rows = support descriptors (M), cols = query descriptors (L).
   Each lane owns one l-column, 8 m-values/tile; top-3 kept in registers; the two
   half-wave row-sets merge via shfl_xor(16) at the end of each l-tile. Main m-loop
   is branch-free (tiles 0..136 fully valid); ragged tile 137 handled as epilogue. */
__global__ void k_sim(const f16* __restrict__ feat, float* __restrict__ out) {
  int bqc = blockIdx.x;                 /* [0,160): (b*NQ+q)*WAY + c */
  int b = bqc / (NQn * WAY);
  int rem = bqc % (NQn * WAY);
  int qi = rem / WAY, cw = rem % WAY;
  int lane = threadIdx.x & 31, wave = threadIdx.x >> 5;
  const f16* qbase = feat + ((long)(b * NQn + qi) * Lloc) * 64;
  const f16* sbase = feat + ((long)(NQIMG + (b * WAY + cw) * SHOT) * Lloc) * 64;
  int kb = (lane >> 4) * 8;
  int db = (lane >> 4) * 8;
  const f16* slane = sbase + (long)(lane & 15) * 64;
  float waveAcc = 0.f;
  for (int lt = wave; lt < 28; lt += 4) {          /* 28 l-tiles of 16 (441 padded) */
    int lcol = lt * 16 + (lane & 15);
    bool vl = lcol < Lloc;
    Frag16 b0, b1;
    if (vl) {
      const uint4* p0 = (const uint4*)(qbase + (long)lcol * 64 + db);
      b0.u[0] = p0[0]; b0.u[1] = p0[2];
      const uint4* p1 = (const uint4*)(qbase + (long)lcol * 64 + 32 + db);
      b1.u[0] = p1[0]; b1.u[1] = p1[2];
    } else {
      b0.u[0] = b0.u[1] = b1.u[0] = b1.u[1] = make_uint4(0, 0, 0, 0);
    }
    float t0 = -3.0e38f, t1 = -3.0e38f, t2 = -3.0e38f;
    const f16* sb = slane;
    for (int mt = 0; mt < 137; ++mt) {             /* fully-valid m-tiles */
      Frag16 a0, a1;
      a0.u[0] = *(const uint4*)(sb + kb);
      a0.u[1] = *(const uint4*)(sb + kb + 16);
      a1.u[0] = *(const uint4*)(sb + 32 + kb);
      a1.u[1] = *(const uint4*)(sb + 32 + kb + 16);
      __builtin_prefetch(sb + 16 * 64, 0, 1);      /* next m-tile A stream */
      v8f cc = {};
      cc = WMMA_F16(a0.h, b0.h, cc);
      cc = WMMA_F16(a1.h, b1.h, cc);
#pragma unroll
      for (int r = 0; r < 8; ++r) top3_insert(cc[r], t0, t1, t2);
      sb += 16 * 64;
    }
    { /* ragged tile 137: rows 2192..2207, valid < 2205 */
      int srow = 137 * 16 + (lane & 15);
      bool vmr = srow < Mtot;
      Frag16 a0, a1;
      if (vmr) {
        a0.u[0] = *(const uint4*)(sb + kb);
        a0.u[1] = *(const uint4*)(sb + kb + 16);
        a1.u[0] = *(const uint4*)(sb + 32 + kb);
        a1.u[1] = *(const uint4*)(sb + 32 + kb + 16);
      } else {
        a0.u[0] = a0.u[1] = a1.u[0] = a1.u[1] = make_uint4(0, 0, 0, 0);
      }
      v8f cc = {};
      cc = WMMA_F16(a0.h, b0.h, cc);
      cc = WMMA_F16(a1.h, b1.h, cc);
      int mrow0 = 137 * 16 + (lane >> 4) * 8;
#pragma unroll
      for (int r = 0; r < 8; ++r)
        if (mrow0 + r < Mtot) top3_insert(cc[r], t0, t1, t2);
    }
    float p0 = __shfl_xor(t0, 16, 32);
    float p1 = __shfl_xor(t1, 16, 32);
    float p2 = __shfl_xor(t2, 16, 32);
    top3_insert(p0, t0, t1, t2);
    top3_insert(p1, t0, t1, t2);
    top3_insert(p2, t0, t1, t2);
    if (lane < 16 && vl) waveAcc += t0 + t1 + t2;
  }
  for (int off = 16; off; off >>= 1) waveAcc += __shfl_down(waveAcc, off, 32);
  __shared__ float wsum[4];
  if (lane == 0) wsum[wave] = waveAcc;
  __syncthreads();
  if (threadIdx.x == 0) out[bqc] = wsum[0] + wsum[1] + wsum[2] + wsum[3];
}

extern "C" void kernel_launch(void* const* d_in, const int* in_sizes, int n_in,
                              void* d_out, int out_size, void* d_ws, size_t ws_size,
                              hipStream_t stream) {
  const float* q  = (const float*)d_in[0];
  const float* s  = (const float*)d_in[1];
  const float* W1 = (const float*)d_in[2];
  const float* g1 = (const float*)d_in[3];
  const float* b1 = (const float*)d_in[4];
  const float* W2 = (const float*)d_in[5];
  const float* g2 = (const float*)d_in[6];
  const float* b2 = (const float*)d_in[7];
  const float* W3 = (const float*)d_in[8];
  const float* g3 = (const float*)d_in[9];
  const float* b3 = (const float*)d_in[10];
  const float* W4 = (const float*)d_in[11];
  const float* g4 = (const float*)d_in[12];
  const float* b4 = (const float*)d_in[13];
  char* ws = (char*)d_ws;
  f16*   raw   = (f16*)(ws + OFF_RAW);
  f16*   act   = (f16*)(ws + OFF_ACT);
  f16*   feat  = (f16*)(ws + OFF_FEAT);
  f16*   wt2   = (f16*)(ws + OFF_WT);
  f16*   wt3   = wt2 + 64 * 576;
  f16*   wt4   = wt3 + 64 * 576;
  float* part0 = (float*)(ws + OFF_PART);
  float* part1 = part0 + STATS_BLOCKS * 64 * 2;
  float* ssb   = (float*)(ws + OFF_SS);
  float* out   = (float*)d_out;

  const size_t LDS_W = 64 * 576 * sizeof(f16);

  /* weight reorder to f16 k-major */
  k_wprep<<<288, 128, 0, stream>>>(W2, wt2);
  k_wprep<<<288, 128, 0, stream>>>(W3, wt3);
  k_wprep<<<288, 128, 0, stream>>>(W4, wt4);

  /* layer 1: conv -> stats -> BN+LReLU+pool (84 -> 42) */
  k_conv1<<<(NIMG * H0 * H0 + 127) / 128, 128, 0, stream>>>(q, s, W1, raw);
  k_stats<<<STATS_BLOCKS, 256, 0, stream>>>(raw, 0L, 32L * H0 * H0, part0);
  k_stats<<<STATS_BLOCKS, 256, 0, stream>>>(raw, 32L * H0 * H0, 50L * H0 * H0, part1);
  k_bnfin<<<1, 128, 0, stream>>>(part0, part1, g1, b1, ssb, 32L * H0 * H0, 50L * H0 * H0);
  k_normpool<<<(NIMG * Hp1 * Hp1 + 127) / 128, 128, 0, stream>>>(raw, act, ssb, H0);

  /* layer 2: WMMA conv (42x42, 111 tiles, 8 waves/block) -> stats -> pool (42 -> 21) */
  k_convw<<<dim3(14, NIMG), 256, LDS_W, stream>>>(act, wt2, raw, Hp1);
  k_stats<<<STATS_BLOCKS, 256, 0, stream>>>(raw, 0L, 32L * Hp1 * Hp1, part0);
  k_stats<<<STATS_BLOCKS, 256, 0, stream>>>(raw, 32L * Hp1 * Hp1, 50L * Hp1 * Hp1, part1);
  k_bnfin<<<1, 128, 0, stream>>>(part0, part1, g2, b2, ssb, 32L * Hp1 * Hp1, 50L * Hp1 * Hp1);
  k_normpool<<<(NIMG * Hp2 * Hp2 + 127) / 128, 128, 0, stream>>>(raw, act, ssb, Hp1);

  /* layer 3: WMMA conv (21x21, 28 tiles), no pool */
  k_convw<<<dim3(4, NIMG), 256, LDS_W, stream>>>(act, wt3, raw, Hp2);
  k_stats<<<STATS_BLOCKS, 256, 0, stream>>>(raw, 0L, 32L * Lloc, part0);
  k_stats<<<STATS_BLOCKS, 256, 0, stream>>>(raw, 32L * Lloc, 50L * Lloc, part1);
  k_bnfin<<<1, 128, 0, stream>>>(part0, part1, g3, b3, ssb, 32L * Lloc, 50L * Lloc);
  k_norm<<<(NIMG * Lloc + 127) / 128, 128, 0, stream>>>(raw, act, ssb);

  /* layer 4: WMMA conv + BN + LReLU + per-descriptor L2 norm */
  k_convw<<<dim3(4, NIMG), 256, LDS_W, stream>>>(act, wt4, raw, Hp2);
  k_stats<<<STATS_BLOCKS, 256, 0, stream>>>(raw, 0L, 32L * Lloc, part0);
  k_stats<<<STATS_BLOCKS, 256, 0, stream>>>(raw, 32L * Lloc, 50L * Lloc, part1);
  k_bnfin<<<1, 128, 0, stream>>>(part0, part1, g4, b4, ssb, 32L * Lloc, 50L * Lloc);
  k_norml2<<<(NIMG * Lloc + 127) / 128, 128, 0, stream>>>(raw, feat, ssb);

  /* fused cosine-sim GEMM + streaming top-3 + reduce: one block per (b,q,c) */
  k_sim<<<BSZ * NQn * WAY, 128, 0, stream>>>(feat, out);
}
